// RecurrentBlock_25391846654027
// MI455X (gfx1250) — matrix operand order repaired and checked by reading, not compiled, not run on hardware
//
#include <hip/hip_runtime.h>
#include <hip/hip_bf16.h>

typedef _Float16 h16;
typedef __attribute__((ext_vector_type(16))) _Float16 v16h;
typedef __attribute__((ext_vector_type(8)))  _Float16 v8h;
typedef __attribute__((ext_vector_type(8)))  float    v8f;
typedef __attribute__((ext_vector_type(4)))  unsigned u32x4;
typedef __attribute__((ext_vector_type(8)))  unsigned u32x8;

static constexpr int  Bc = 4, Tc = 8192, Wc = 1024, Lc = 1024, Hc = 8, TWc = 4, Dc = 128;
static constexpr long MT = (long)Bc * Tc;        // 32768 GEMM rows
static constexpr long ME = MT * (long)Lc;        // elements in [B,T,L]
static constexpr int  NC = 64;                   // scan chunks per sequence
static constexpr int  CHL = Tc / NC;             // chunk length (128)
static constexpr float A_SCALEc = 8.0f;

#define TILE_M 128
#define TILE_N 128
#define TILE_K 64
#define LDS_STRIDE 72   // halves; 144B rows (128B data + 16B TDM pad) -> conflict-free

enum { ACT_NONE = 0, ACT_GELU_F16 = 1, ACT_SIGMOID_F32 = 2 };

__device__ __forceinline__ float gelu_tanh(float x) {
  float inner = 0.7978845608028654f * (x + 0.044715f * x * x * x);
  return 0.5f * x * (1.0f + tanhf(inner));
}

// CDNA5 Tensor Data Mover: DMA a 2D tile (128 rows x 64 halves, row stride lda
// elements) from global to LDS, inserting 16B padding every 128B so LDS rows
// land at stride 72 halves. Descriptor per ISA 8.3-8.4; 2-group form (<=2D).
__device__ __forceinline__ void tdm_load_tile(unsigned lds_addr, const h16* gptr,
                                              unsigned lda_elems) {
  unsigned long ga = (unsigned long)gptr;
  u32x4 g0;
  g0[0] = 1u;                                   // count=1 (valid), no gather
  g0[1] = lds_addr;                             // lds_addr[31:0] (bytes)
  g0[2] = (unsigned)ga;                         // global_addr[31:0]
  g0[3] = (unsigned)(ga >> 32) | (2u << 30);    // global_addr[56:32] | type=2
  u32x8 g1;
  g1[0] = (1u << 16)                            // data_size = 1 -> 2 bytes
        | (1u << 20)                            // pad_enable
        | (4u << 22)                            // pad_interval: 32 DWORDs (128B)
        | (3u << 25);                           // pad_amount: 4 DWORDs (16B)
  g1[1] = (unsigned)TILE_K << 16;               // tensor_dim0 = 64 (lo16)
  g1[2] = (unsigned)TILE_M << 16;               // tensor_dim0 hi=0 | tensor_dim1 lo16 = 128
  g1[3] = (unsigned)TILE_K << 16;               // tensor_dim1 hi=0 | tile_dim0 = 64
  g1[4] = (unsigned)TILE_M;                     // tile_dim1 = 128 | tile_dim2 = 0
  g1[5] = lda_elems;                            // tensor_dim0_stride[31:0]
  g1[6] = 0u;                                   // stride hi | dim1_stride lo
  g1[7] = 0u;
  asm volatile("tensor_load_to_lds %0, %1, null, null"
               :: "s"(g0), "s"(g1)
               : "memory");
}

// C = act(A[M,K] @ Bt[N,K]^T + bias), A/Bt f16 row-major, fp32 acc via WMMA.
// Double-buffered LDS tiles staged by the Tensor Data Mover (wave 0 issues,
// TENSORcnt-ordered, barrier publishes to the workgroup).
template <int ACT>
__global__ __launch_bounds__(256) void gemm_f16_kernel(
    const h16* __restrict__ A, long lda, long aBatch,
    const h16* __restrict__ Bt, long ldb, long bBatch,
    const float* __restrict__ bias, long biasBatch,
    float* __restrict__ Cf, h16* __restrict__ Ch, long ldc, long cBatch,
    int K) {
  __shared__ __align__(16) h16 As[2][TILE_M * LDS_STRIDE];
  __shared__ __align__(16) h16 Bs[2][TILE_N * LDS_STRIDE];

  const int tid  = threadIdx.x;
  const int lane = tid & 31;
  const int wave = tid >> 5;
  const int wm   = wave & 3;   // 4 wave-rows of 32
  const int wn   = wave >> 2;  // 2 wave-cols of 64
  const long m0  = (long)blockIdx.y * TILE_M;
  const long n0  = (long)blockIdx.x * TILE_N;
  const int  bat = blockIdx.z;

  const h16* Ab = A  + (long)bat * aBatch;
  const h16* Bb = Bt + (long)bat * bBatch;

  v8f acc[2][4] = {};

  const int row_in = lane & 15;
  const int a_koff = (lane >> 4) * 8;    // A frag: K in {koff..+7} U {koff+16..+23}
  const int b_koff = (lane >> 4) * 8;    // B frag: K in {koff..+7} U {koff+16..+23}

  auto issue_tile = [&](int kb, int buf) {
    tdm_load_tile((unsigned)(size_t)&As[buf][0], Ab + m0 * lda + kb, (unsigned)lda);
    tdm_load_tile((unsigned)(size_t)&Bs[buf][0], Bb + n0 * ldb + kb, (unsigned)ldb);
  };

  const int nt = K / TILE_K;
  if (wave == 0) issue_tile(0, 0);

  for (int it = 0; it < nt; ++it) {
    const int cur = it & 1;
    __syncthreads();  // WAR: everyone done reading buf[cur^1] from previous iter
    if (wave == 0) {
      if (it + 1 < nt) {
        issue_tile((it + 1) * TILE_K, cur ^ 1);
        __builtin_amdgcn_s_wait_tensorcnt(2);  // TDM in-order: tile it landed
      } else {
        __builtin_amdgcn_s_wait_tensorcnt(0);
      }
    }
    __syncthreads();  // tile-it data visible to all waves

#pragma unroll
    for (int ks = 0; ks < TILE_K; ks += 32) {
      v16h afrag[2], bfrag[4];
#pragma unroll
      for (int i = 0; i < 2; ++i) {
        const h16* p = &As[cur][(wm * 32 + i * 16 + row_in) * LDS_STRIDE + ks + a_koff];
        v8h lo = *(const v8h*)p;
        v8h hi = *(const v8h*)(p + 16);
        afrag[i] = __builtin_shufflevector(lo, hi, 0,1,2,3,4,5,6,7,8,9,10,11,12,13,14,15);
      }
#pragma unroll
      for (int j = 0; j < 4; ++j) {
        const h16* p = &Bs[cur][(wn * 64 + j * 16 + row_in) * LDS_STRIDE + ks + b_koff];
        v8h lo = *(const v8h*)p;
        v8h hi = *(const v8h*)(p + 16);
        bfrag[j] = __builtin_shufflevector(lo, hi, 0,1,2,3,4,5,6,7,8,9,10,11,12,13,14,15);
      }
#pragma unroll
      for (int i = 0; i < 2; ++i)
#pragma unroll
        for (int j = 0; j < 4; ++j)
          acc[i][j] = __builtin_amdgcn_wmma_f32_16x16x32_f16(
              false, afrag[i], false, bfrag[j], (short)0, acc[i][j], false, false);
    }
  }

  const float* bp = bias + (long)bat * biasBatch;
#pragma unroll
  for (int i = 0; i < 2; ++i)
#pragma unroll
    for (int j = 0; j < 4; ++j)
#pragma unroll
      for (int r = 0; r < 8; ++r) {
        int  mrow = wm * 32 + i * 16 + ((lane < 16) ? r : (r + 8));
        int  ncol = wn * 64 + j * 16 + (lane & 15);
        long gm = m0 + mrow;
        long gn = n0 + ncol;
        float v = acc[i][j][r] + bp[gn];
        long cidx = (long)bat * cBatch + gm * ldc + gn;
        if (ACT == ACT_GELU_F16)        Ch[cidx] = (h16)gelu_tanh(v);
        else if (ACT == ACT_SIGMOID_F32) Cf[cidx] = 1.0f / (1.0f + expf(-v));
        else                             Cf[cidx] = v;
      }
}

// out[N,K] (f16) = in[K,N]^T (f32); batched via blockIdx.z
__global__ __launch_bounds__(256) void transpose_f32_f16(
    const float* __restrict__ in, h16* __restrict__ out, int rowsIn, int colsIn) {
  long n = (long)rowsIn * colsIn;
  long i = (long)blockIdx.x * 256 + threadIdx.x;
  if (i >= n) return;
  const float* ib = in + (long)blockIdx.z * n;
  h16* ob = out + (long)blockIdx.z * n;
  long k = i % rowsIn;
  long c = i / rowsIn;
  ob[i] = (h16)ib[k * colsIn + c];
}

__global__ __launch_bounds__(256) void f32_to_f16_kernel(
    const float* __restrict__ in, h16* __restrict__ out, long n) {
  long i = (long)blockIdx.x * 256 + threadIdx.x;
  if (i < n) out[i] = (h16)in[i];
}

// causal depthwise conv1d (width 4, segment-aware); emits f32 and f16 copies
__global__ __launch_bounds__(256) void conv_kernel(
    const float* __restrict__ u, const int* __restrict__ pos,
    const float* __restrict__ cw, const float* __restrict__ cb,
    float* __restrict__ uc, h16* __restrict__ uch) {
  long i = (long)blockIdx.x * 256 + threadIdx.x;
  if (i >= ME) return;
  int  l  = (int)(i % Lc);
  long bt = i / Lc;
  int  t  = (int)(bt % Tc);
  int  b  = (int)(bt / Tc);
  int  p  = pos[(long)b * Tc + t];
  float acc = cb[l];
#pragma unroll
  for (int k = 0; k < TWc; ++k) {
    int shift = TWc - 1 - k;
    if (p >= shift && t >= shift)
      acc += u[i - (long)shift * Lc] * cw[k * Lc + l];
  }
  uc[i] = acc;
  uch[i] = (h16)acc;
}

// in-place: gx(sigmoid gate_x) -> normed_x ; ga(sigmoid gate_a) -> a_m
__global__ __launch_bounds__(256) void prep_kernel(
    const float* __restrict__ uc, const int* __restrict__ pos,
    const float* __restrict__ a_param,
    float* __restrict__ gx, float* __restrict__ ga) {
  long i = (long)blockIdx.x * 256 + threadIdx.x;
  if (i >= ME) return;
  int  l  = (int)(i % Lc);
  long bt = i / Lc;
  int  t  = (int)(bt % Tc);
  int  b  = (int)(bt / Tc);
  float ap = a_param[l];
  float sp = (ap > 20.0f) ? ap : log1pf(expf(ap));   // softplus
  float la = -A_SCALEc * ga[i] * sp;
  float a  = expf(la);
  float mult = sqrtf(fmaxf(0.0f, 1.0f - expf(2.0f * la)));
  bool reset = (pos[(long)b * Tc + t] == 0);
  float nx = uc[i] * gx[i] * (reset ? 1.0f : mult);
  gx[i] = nx;
  ga[i] = reset ? 0.0f : a;
}

// chunked linear-recurrence scan: h_t = a_t*h_{t-1} + x_t
__global__ __launch_bounds__(256) void scan_pass1(
    const float* __restrict__ am, const float* __restrict__ nx,
    float* __restrict__ cA, float* __restrict__ cB) {
  long tid = (long)blockIdx.x * 256 + threadIdx.x;
  if (tid >= (long)Bc * NC * Lc) return;
  int  l = (int)(tid % Lc);
  long r = tid / Lc;
  int  c = (int)(r % NC);
  int  b = (int)(r / NC);
  long base = ((long)b * Tc + (long)c * CHL) * Lc + l;
  float A = 1.0f, Bv = 0.0f;
  for (int t = 0; t < CHL; ++t) {
    float a = am[base], xv = nx[base];
    A  = a * A;
    Bv = a * Bv + xv;
    base += Lc;
  }
  cA[tid] = A;
  cB[tid] = Bv;
}

__global__ __launch_bounds__(256) void scan_pass2(
    const float* __restrict__ cA, const float* __restrict__ cB,
    float* __restrict__ carry) {
  long tid = (long)blockIdx.x * 256 + threadIdx.x;
  if (tid >= (long)Bc * Lc) return;
  int l = (int)(tid % Lc);
  int b = (int)(tid / Lc);
  float hc = 0.0f;
  for (int c = 0; c < NC; ++c) {
    long i = ((long)b * NC + c) * Lc + l;
    carry[i] = hc;
    hc = cA[i] * hc + cB[i];
  }
}

__global__ __launch_bounds__(256) void scan_pass3(
    const float* __restrict__ am, const float* __restrict__ nx,
    const float* __restrict__ carry, float* __restrict__ h) {
  long tid = (long)blockIdx.x * 256 + threadIdx.x;
  if (tid >= (long)Bc * NC * Lc) return;
  int  l = (int)(tid % Lc);
  long r = tid / Lc;
  int  c = (int)(r % NC);
  int  b = (int)(r / NC);
  long base = ((long)b * Tc + (long)c * CHL) * Lc + l;
  float hv = carry[tid];
  for (int t = 0; t < CHL; ++t) {
    hv = am[base] * hv + nx[base];
    h[base] = hv;
    base += Lc;
  }
}

// g = (f16)(h * y)
__global__ __launch_bounds__(256) void mul_convert_kernel(
    const float* __restrict__ h, const h16* __restrict__ y,
    h16* __restrict__ g, long n) {
  long i = (long)blockIdx.x * 256 + threadIdx.x;
  if (i < n) g[i] = (h16)(h[i] * (float)y[i]);
}

extern "C" void kernel_launch(void* const* d_in, const int* in_sizes, int n_in,
                              void* d_out, int out_size, void* d_ws, size_t ws_size,
                              hipStream_t stream) {
  (void)in_sizes; (void)n_in; (void)out_size; (void)ws_size;
  const float* x       = (const float*)d_in[0];
  const int*   segpos  = (const int*)d_in[1];
  const float* Wy      = (const float*)d_in[2];
  const float* by      = (const float*)d_in[3];
  const float* Wx      = (const float*)d_in[4];
  const float* bx      = (const float*)d_in[5];
  const float* conv_w  = (const float*)d_in[6];
  const float* conv_b  = (const float*)d_in[7];
  const float* a_param = (const float*)d_in[8];
  const float* igw     = (const float*)d_in[9];
  const float* igb     = (const float*)d_in[10];
  const float* agw     = (const float*)d_in[11];
  const float* agb     = (const float*)d_in[12];
  const float* Wout    = (const float*)d_in[13];
  const float* bout    = (const float*)d_in[14];
  float* out = (float*)d_out;

  char* ws = (char*)d_ws;
  size_t off = 0;
  auto alloc = [&](size_t bytes) -> void* {
    void* p = ws + off;
    off += (bytes + 255) & ~(size_t)255;
    return p;
  };

  h16*   bufh0 = (h16*)alloc((size_t)ME * sizeof(h16));            // xh -> gh
  h16*   Wyt   = (h16*)alloc((size_t)Lc * Wc * sizeof(h16));       // [L,W]
  h16*   Wxt   = (h16*)alloc((size_t)Lc * Wc * sizeof(h16));       // [L,W]
  h16*   Woutt = (h16*)alloc((size_t)Wc * Lc * sizeof(h16));       // [W,L]
  h16*   gwt   = (h16*)alloc((size_t)2 * Hc * Dc * Dc * sizeof(h16)); // igate^T, agate^T
  h16*   yh    = (h16*)alloc((size_t)ME * sizeof(h16));            // gelu branch, f16
  h16*   uch   = (h16*)alloc((size_t)ME * sizeof(h16));            // post-conv u, f16
  float* upre  = (float*)alloc((size_t)ME * sizeof(float));        // x@Wx -> later h
  float* uconv = (float*)alloc((size_t)ME * sizeof(float));        // post-conv u
  float* gx    = (float*)alloc((size_t)ME * sizeof(float));        // gate_x -> normed_x
  float* ga    = (float*)alloc((size_t)ME * sizeof(float));        // gate_a -> a_m
  float* cA    = (float*)alloc((size_t)Bc * NC * Lc * sizeof(float));
  float* cB    = (float*)alloc((size_t)Bc * NC * Lc * sizeof(float));
  float* carry = (float*)alloc((size_t)Bc * NC * Lc * sizeof(float));

  dim3 blk(256);
  auto eb = [](long n) { return dim3((unsigned)((n + 255) / 256)); };

  // weight transposes + input conversion
  transpose_f32_f16<<<eb((long)Wc * Lc), blk, 0, stream>>>(Wy, Wyt, Wc, Lc);
  transpose_f32_f16<<<eb((long)Wc * Lc), blk, 0, stream>>>(Wx, Wxt, Wc, Lc);
  transpose_f32_f16<<<eb((long)Lc * Wc), blk, 0, stream>>>(Wout, Woutt, Lc, Wc);
  transpose_f32_f16<<<dim3((unsigned)((Dc * Dc + 255) / 256), 1, Hc), blk, 0, stream>>>(
      igw, gwt, Dc, Dc);
  transpose_f32_f16<<<dim3((unsigned)((Dc * Dc + 255) / 256), 1, Hc), blk, 0, stream>>>(
      agw, gwt + (size_t)Hc * Dc * Dc, Dc, Dc);
  f32_to_f16_kernel<<<eb(ME), blk, 0, stream>>>(x, bufh0, ME);

  // y = gelu(x@Wy + by)  (f16 out);  u_pre = x@Wx + bx (f32 out)
  gemm_f16_kernel<ACT_GELU_F16><<<dim3(Lc / TILE_N, MT / TILE_M, 1), blk, 0, stream>>>(
      bufh0, Wc, 0, Wyt, Wc, 0, by, 0, nullptr, yh, Lc, 0, Wc);
  gemm_f16_kernel<ACT_NONE><<<dim3(Lc / TILE_N, MT / TILE_M, 1), blk, 0, stream>>>(
      bufh0, Wc, 0, Wxt, Wc, 0, bx, 0, upre, nullptr, Lc, 0, Wc);

  // causal depthwise conv (emits f32 + f16)
  conv_kernel<<<eb(ME), blk, 0, stream>>>(upre, segpos, conv_w, conv_b, uconv, uch);

  // block-diagonal gate GEMMs (batched over heads) with fused sigmoid
  gemm_f16_kernel<ACT_SIGMOID_F32><<<dim3(1, MT / TILE_M, Hc), blk, 0, stream>>>(
      uch, Lc, Dc, gwt, Dc, (long)Dc * Dc, igb, Dc, gx, nullptr, Lc, Dc, Dc);
  gemm_f16_kernel<ACT_SIGMOID_F32><<<dim3(1, MT / TILE_M, Hc), blk, 0, stream>>>(
      uch, Lc, Dc, gwt + (size_t)Hc * Dc * Dc, Dc, (long)Dc * Dc, agb, Dc, ga, nullptr, Lc, Dc, Dc);

  // RG-LRU elementwise prep (in-place into gx/ga)
  prep_kernel<<<eb(ME), blk, 0, stream>>>(uconv, segpos, a_param, gx, ga);

  // chunked linear scan -> h (reuses upre)
  scan_pass1<<<eb((long)Bc * NC * Lc), blk, 0, stream>>>(ga, gx, cA, cB);
  scan_pass2<<<eb((long)Bc * Lc), blk, 0, stream>>>(cA, cB, carry);
  scan_pass3<<<eb((long)Bc * NC * Lc), blk, 0, stream>>>(ga, gx, carry, upre);

  // g = h * y  (f16, reuses bufh0) then final GEMM with bias
  mul_convert_kernel<<<eb(ME), blk, 0, stream>>>(upre, yh, bufh0, ME);
  gemm_f16_kernel<ACT_NONE><<<dim3(Wc / TILE_N, MT / TILE_M, 1), blk, 0, stream>>>(
      bufh0, Lc, 0, Woutt, Lc, 0, bout, 0, out, nullptr, Wc, 0, Lc);
}
